// ScaleDotProductAttention_79663053406595
// MI455X (gfx1250) — matrix operand order repaired and checked by reading, not compiled, not run on hardware
//
#include <hip/hip_runtime.h>

// ---------------------------------------------------------------------------
// Scaled dot-product attention, MI455X (gfx1250), wave32 + WMMA f16.
//   scores = (Q K^T) * D^-0.5 ; masked -> 1e-12 (reference quirk, NOT -inf)
//   attn   = softmax(scores)   (materialized to d_out, 1 GB -> NT stores)
//   out    = attn @ V
// One wave handles one 16-row query tile; 2 waves per block.
// ---------------------------------------------------------------------------

typedef __attribute__((ext_vector_type(16))) _Float16 v16h;
typedef __attribute__((ext_vector_type(8)))  _Float16 v8h;
typedef __attribute__((ext_vector_type(4)))  _Float16 v4h;
typedef __attribute__((ext_vector_type(8)))  float    v8f;
typedef __attribute__((ext_vector_type(4)))  float    v4f;

#define B_  4
#define H_  16
#define S_  2048
#define D_  64
#define BH_ (B_ * H_)

// LDS: per-wave 16 rows x 2056 halfs (stride 4112B: 16B aligned, bank-skewed)
#define ROWH   2056
#define WLDS   (16 * ROWH * 2)       // 65792 bytes per wave
#define INVOFF (2 * WLDS)            // inv[16] per wave after the two P tiles

__global__ void __launch_bounds__(64)
sdpa_wmma_kernel(const float* __restrict__ q, const float* __restrict__ k,
                 const float* __restrict__ v, const int* __restrict__ mask,
                 float* __restrict__ outO, float* __restrict__ outA)
{
    extern __shared__ char smem[];
    const int tid    = threadIdx.x;
    const int wave   = tid >> 5;
    const int lane   = tid & 31;
    const int laneLo = lane & 15;     // N / M-low index
    const int hi     = lane >> 4;     // lane-half selector in WMMA layouts

    const int tile = blockIdx.x * 2 + wave;       // 8192 tiles total
    const int bh   = tile >> 7;                   // 128 row-tiles per (b,h)
    const int row0 = (tile & 127) * 16;
    const int b    = bh >> 4;

    _Float16* Pw   = (_Float16*)(smem + wave * WLDS);
    float*    invw = (float*)(smem + INVOFF + wave * 64);

    const float scale  = 0.125f;                  // 1/sqrt(64)
    const size_t base  = (size_t)bh * S_ * D_;

    // ---- Stage Q tile into WMMA A-layout (f16 16x32: 8-contig runs) ------
    // A element 2v+e  <->  d = c*32 + (v>>2)*16 + hi*8 + (v&3)*2 + e
    const float* qRow = q + base + (size_t)(row0 + laneLo) * D_;
    v16h aQ[2];
#pragma unroll
    for (int c = 0; c < 2; ++c) {
#pragma unroll
        for (int g = 0; g < 2; ++g) {
            const float* p = qRow + c * 32 + g * 16 + hi * 8;
            v4f f0 = *(const v4f*)(p);
            v4f f1 = *(const v4f*)(p + 4);
#pragma unroll
            for (int t = 0; t < 4; ++t) {
                aQ[c][g * 8 + t]     = (_Float16)f0[t];
                aQ[c][g * 8 + 4 + t] = (_Float16)f1[t];
            }
        }
    }

    // ---- Pass 1: S = QK^T * scale, mask-fill, f16 scores -> LDS ----------
    // B element t (0..15) <-> k = (t>>3)*16 + hi*8 + (t&7)  (two 8-float runs of K row)
    const int* maskB = mask + b * S_;
#pragma unroll 1
    for (int j = 0; j < 128; ++j) {
        const int   col  = j * 16 + laneLo;           // tile col == K row idx
        const float* kRow = k + base + (size_t)col * D_;
        __builtin_prefetch(kRow + 16 * D_, 0, 1);     // next K tile -> global_prefetch
        v8f acc = {};
#pragma unroll
        for (int c = 0; c < 2; ++c) {
            const float* p = kRow + c * 32 + hi * 8;
            v4f f0 = *(const v4f*)(p);
            v4f f1 = *(const v4f*)(p + 4);
            v4f f2 = *(const v4f*)(p + 16);
            v4f f3 = *(const v4f*)(p + 20);
            v16h bK;
#pragma unroll
            for (int t = 0; t < 4; ++t) {
                bK[t]      = (_Float16)f0[t];
                bK[4 + t]  = (_Float16)f1[t];
                bK[8 + t]  = (_Float16)f2[t];
                bK[12 + t] = (_Float16)f3[t];
            }
            acc = __builtin_amdgcn_wmma_f32_16x16x32_f16(
                      false, aQ[c], false, bK, (short)0, acc, false, false);
        }
        const bool live = (maskB[col] != 0);
#pragma unroll
        for (int r = 0; r < 8; ++r) {                 // D row m = r + 8*hi
            float sv = live ? acc[r] * scale : 1e-12f;
            Pw[(r + 8 * hi) * ROWH + col] = (_Float16)sv;
        }
    }
    __syncthreads();

    // ---- Pass 2: softmax (2 lanes per row, 1024 cols each) ---------------
    {
        const int row = laneLo;
        _Float16* prow = Pw + row * ROWH + hi * 1024;
        float mx = -3.0e38f;
#pragma unroll 1
        for (int i = 0; i < 128; ++i) {
            v8h hv = *(const v8h*)(prow + i * 8);
#pragma unroll
            for (int t = 0; t < 8; ++t) mx = fmaxf(mx, (float)hv[t]);
        }
        mx = fmaxf(mx, __shfl_xor(mx, 16, 32));
        float sum = 0.f;
#pragma unroll 1
        for (int i = 0; i < 128; ++i) {
            v8h hv = *(const v8h*)(prow + i * 8);
            v8h ev;
#pragma unroll
            for (int t = 0; t < 8; ++t) {
                float e = __expf((float)hv[t] - mx);
                sum += e;
                ev[t] = (_Float16)e;
            }
            *(v8h*)(prow + i * 8) = ev;               // unnormalized P in LDS
        }
        sum += __shfl_xor(sum, 16, 32);
        if (hi == 0) invw[row] = 1.0f / sum;
    }
    __syncthreads();

    // ---- Pass 2b: stream normalized attention out (coalesced, NT) -------
    float* attnBase = outA + ((size_t)bh * S_ + row0) * (size_t)S_;
#pragma unroll 1
    for (int t = 0; t < 256; ++t) {
        const int flat = t * 128 + lane * 4;          // 128 floats / iter
        const int rowT = t >> 4;                      // uniform per iter
        const int col  = flat & (S_ - 1);
        const float is = invw[rowT];
        v4h h4 = *(const v4h*)(Pw + rowT * ROWH + col);
        v4f o;
#pragma unroll
        for (int e = 0; e < 4; ++e) o[e] = (float)h4[e] * is;
        __builtin_nontemporal_store(o, (v4f*)(attnBase + (size_t)rowT * S_ + col));
    }

    // ---- Pass 3: O = P @ V (normalize at epilogue) -----------------------
    v8f accO[4] = {{}, {}, {}, {}};
    const _Float16* pA  = Pw + laneLo * ROWH;
    const float*    vC  = v + base + laneLo;          // + row*64 + nt*16 later
#pragma unroll 1
    for (int c = 0; c < 64; ++c) {
        v16h aP;
#pragma unroll
        for (int g = 0; g < 2; ++g) {
            v8h hv = *(const v8h*)(pA + c * 32 + g * 16 + hi * 8);
#pragma unroll
            for (int t = 0; t < 8; ++t) aP[g * 8 + t] = hv[t];
        }
        const float* vRow = vC + (size_t)(c * 32 + hi * 8) * D_;
#pragma unroll
        for (int nt = 0; nt < 4; ++nt) {
            v16h bV;
#pragma unroll
            for (int t = 0; t < 16; ++t)              // k = (t>>3)*16 + hi*8 + (t&7)
                bV[t] = (_Float16)vRow[(size_t)((t & 7) + (t >> 3) * 16) * D_ + nt * 16];
            accO[nt] = __builtin_amdgcn_wmma_f32_16x16x32_f16(
                           false, aP, false, bV, (short)0, accO[nt], false, false);
        }
    }

    float invr[8];
#pragma unroll
    for (int r = 0; r < 8; ++r) invr[r] = invw[r + 8 * hi];
    float* oBase = outO + ((size_t)bh * S_ + row0) * D_;
#pragma unroll
    for (int nt = 0; nt < 4; ++nt)
#pragma unroll
        for (int r = 0; r < 8; ++r)
            oBase[(size_t)(r + 8 * hi) * D_ + nt * 16 + laneLo] =
                accO[nt][r] * invr[r];
}

extern "C" void kernel_launch(void* const* d_in, const int* in_sizes, int n_in,
                              void* d_out, int out_size, void* d_ws, size_t ws_size,
                              hipStream_t stream) {
    const float* q    = (const float*)d_in[0];
    const float* k    = (const float*)d_in[1];
    const float* v    = (const float*)d_in[2];
    const int*   mask = (const int*)d_in[3];
    float* outO = (float*)d_out;
    float* outA = outO + (size_t)B_ * H_ * S_ * D_;   // tuple: (output, attention)

    const size_t smemBytes = INVOFF + 2 * 64;         // 131,712 B (< 320 KB WGP pool)
    dim3 grid(BH_ * (S_ / 16) / 2);                   // 4096 blocks x 2 waves
    dim3 block(64);
    sdpa_wmma_kernel<<<grid, block, smemBytes, stream>>>(q, k, v, mask, outO, outA);
}
